// CLLoss_58789512348275
// MI455X (gfx1250) — compile-verified, matrix operand order checked by reading, not run on hardware
//
#include <hip/hip_runtime.h>

#define BSZ   8192
#define CDIM  512
#define TPOS  8
#define TNEG  64
#define WAVES 8
#define NT    (BSZ / 16)        // 512 column tiles
#define BSTRIDE 520             // padded LDS row stride (halves): 1040B -> bank spread
#define KSTEPS (CDIM / 32)      // 16 WMMA K-steps
#define INVTEMP 2.0f            // 1 / 0.5

typedef __attribute__((ext_vector_type(16))) _Float16 v16h;
typedef __attribute__((ext_vector_type(8)))  float    v8f;
typedef __attribute__((ext_vector_type(4)))  float    v4f;

union Frag { v16h h; v4f f[2]; };

// ---------------- Kernel 1: L2 normalize rows, emit f16 ----------------
__global__ __launch_bounds__(256) void cl_normalize_kernel(
    const float* __restrict__ x, _Float16* __restrict__ nf) {
  const int w    = threadIdx.x >> 5;
  const int lane = threadIdx.x & 31;
  const int row  = blockIdx.x * WAVES + w;

  const float* rp = x + (size_t)row * CDIM + lane * 16;
  v4f a0 = *(const v4f*)(rp);
  v4f a1 = *(const v4f*)(rp + 4);
  v4f a2 = *(const v4f*)(rp + 8);
  v4f a3 = *(const v4f*)(rp + 12);

  float s = 0.f;
#pragma unroll
  for (int i = 0; i < 4; i++)
    s += a0[i]*a0[i] + a1[i]*a1[i] + a2[i]*a2[i] + a3[i]*a3[i];
#pragma unroll
  for (int off = 16; off > 0; off >>= 1) s += __shfl_xor(s, off, 32);

  const float inv = 1.0f / fmaxf(sqrtf(s), 1e-12f);

  v16h h;
#pragma unroll
  for (int i = 0; i < 4; i++) {
    h[i]      = (_Float16)(a0[i] * inv);
    h[4 + i]  = (_Float16)(a1[i] * inv);
    h[8 + i]  = (_Float16)(a2[i] * inv);
    h[12 + i] = (_Float16)(a3[i] * inv);
  }
  *(v16h*)(nf + (size_t)row * CDIM + lane * 16) = h;
}

// async-copy one 16B chunk: global -> LDS (ASYNCcnt-tracked)
__device__ __forceinline__ void async_b128(const _Float16* g, const _Float16* l) {
  unsigned lds = (unsigned)(size_t)(const void*)l;  // low 32b of generic LDS ptr = LDS offset
  asm volatile("global_load_async_to_lds_b128 %0, %1, off"
               :: "v"(lds), "v"(g) : "memory");
}

// cooperative async stage of one B panel (16 rows x 512 halves, contiguous in mem)
__device__ __forceinline__ void issue_panel(const _Float16* __restrict__ nf,
                                            _Float16* pan, int CB, int tid) {
#pragma unroll
  for (int i = 0; i < 4; i++) {
    int s   = tid + i * 256;     // slot 0..1023, 8 halves each
    int row = s >> 6;            // 64 slots per 512-half row
    int col = (s & 63) * 8;
    async_b128(nf + (size_t)(CB + row) * CDIM + col,
               pan + row * BSTRIDE + col);
  }
}

// ------- Kernel 2: async-staged WMMA GEMM + streaming top-k + fused loss -------
__global__ __launch_bounds__(256) void cl_mine_loss_kernel(
    const _Float16* __restrict__ nf,
    const int* __restrict__ tgt,
    float* __restrict__ out) {
  __shared__ __attribute__((aligned(16))) _Float16 bpanel[2][16 * BSTRIDE]; // 2x16.25KB
  __shared__ float simbuf[WAVES][16][17];
  __shared__ float poslist[WAVES][16][TPOS];
  __shared__ float neglist[WAVES][16][TNEG];
  __shared__ int   ctarg[WAVES][16];

  const float INF = __builtin_inff();

  const int tid  = threadIdx.x;
  const int w    = tid >> 5;
  const int lane = tid & 31;
  const int lr   = lane & 15;
  const int hi   = lane >> 4;
  const int R    = (blockIdx.x * WAVES + w) * 16;   // query row base

  if (lane < 16) {
#pragma unroll
    for (int p = 0; p < TPOS; p++) poslist[w][lane][p] = INF;
  } else {
#pragma unroll
    for (int k = 0; k < TNEG; k++) neglist[w][lane - 16][k] = -INF;
  }
  const int myclass = tgt[R + lr];

  float best = (lane < 16) ? INF : -INF;
  int   bidx = 0;

  // explicit A-tile hoist: 16 fragments live in VGPRs for the whole column loop
  const _Float16* arow = nf + (size_t)(R + lr) * CDIM + hi * 8;
  Frag afrag[KSTEPS];
#pragma unroll
  for (int ks = 0; ks < KSTEPS; ks++) {
    afrag[ks].f[0] = *(const v4f*)(arow + ks * 32);
    afrag[ks].f[1] = *(const v4f*)(arow + ks * 32 + 16);
  }

  issue_panel(nf, &bpanel[0][0], 0, tid);   // prologue: panel 0 in flight

  for (int t = 0; t < NT; t++) {
    const int buf = t & 1;

    asm volatile("s_wait_asynccnt 0x0" ::: "memory");  // this wave's panel t landed
    __syncthreads();   // all waves' portions landed; all waves done with prev iter

    if (t + 1 < NT) issue_panel(nf, &bpanel[buf ^ 1][0], (t + 1) * 16, tid);
    if (lane < 16) ctarg[w][lane] = tgt[t * 16 + lane];

    const _Float16* bbase = &bpanel[buf][lr * BSTRIDE + hi * 8];

    // software-pipelined K-loop: B fragments double-buffered in registers
    Frag bb[2];
    bb[0].f[0] = *(const v4f*)(bbase);
    bb[0].f[1] = *(const v4f*)(bbase + 16);

    v8f acc = {};
#pragma unroll
    for (int ks = 0; ks < KSTEPS; ks++) {
      if (ks + 1 < KSTEPS) {   // prefetch next B pair before this WMMA
        bb[(ks + 1) & 1].f[0] = *(const v4f*)(bbase + (ks + 1) * 32);
        bb[(ks + 1) & 1].f[1] = *(const v4f*)(bbase + (ks + 1) * 32 + 16);
      }
      acc = __builtin_amdgcn_wmma_f32_16x16x32_f16(
          false, afrag[ks].h, false, bb[ks & 1].h, (short)0, acc, false, false);
    }

    // C layout: acc[r] = sim[M = r + 8*hi][N = lr]
#pragma unroll
    for (int r = 0; r < 8; r++) simbuf[w][r + hi * 8][lr] = acc[r];
    __syncthreads();

    if (lane < 16) {               // positives: smallest-8 among same class
      const int row = lane;
#pragma unroll 4
      for (int j = 0; j < 16; j++) {
        float v = simbuf[w][row][j];
        if (ctarg[w][j] == myclass && v < best) {
          poslist[w][row][bidx] = v;
          float m = poslist[w][row][0]; int mi = 0;
#pragma unroll
          for (int p = 1; p < TPOS; p++) {
            float tv = poslist[w][row][p];
            if (tv > m) { m = tv; mi = p; }
          }
          best = m; bidx = mi;
        }
      }
    } else {                       // negatives: largest-64 among other class
      const int row = lane - 16;
#pragma unroll 4
      for (int j = 0; j < 16; j++) {
        float v = simbuf[w][row][j];
        if (ctarg[w][j] != myclass && v > best) {
          neglist[w][row][bidx] = v;
          float m = neglist[w][row][0]; int mi = 0;
          for (int k = 1; k < TNEG; k++) {
            float tv = neglist[w][row][k];
            if (tv < m) { m = tv; mi = k; }
          }
          best = m; bidx = mi;
        }
      }
    }
  }
  __syncthreads();

  // -------- fused loss: -log_softmax([pos_p, neg_0..63]/T)[0], mean over p
  if (lane < 16) {
    const int row = lane;

    float mneg = -INF;
#pragma unroll
    for (int k = 0; k < TNEG; k++) mneg = fmaxf(mneg, neglist[w][row][k]);

    float mnegT = -INF;
    float S = 0.f;
    if (mneg > -INF) {
      mnegT = mneg * INVTEMP;
#pragma unroll
      for (int k = 0; k < TNEG; k++) {
        float xn = neglist[w][row][k] * INVTEMP;
        S += __expf(xn - mnegT);
      }
    }

    float lsum = 0.f;
#pragma unroll
    for (int p = 0; p < TPOS; p++) {
      float pv = poslist[w][row][p];
      float xp = (pv == INF) ? -INF : pv * INVTEMP;  // reference: +inf -> -inf
      float m  = fmaxf(xp, mnegT);
      float l;
      if (m == -INF) {
        l = INF;  // empty logits -> non-finite -> dropped
      } else {
        float sum = ((xp == -INF) ? 0.f : __expf(xp - m)) +
                    ((mnegT == -INF) ? 0.f : S * __expf(mnegT - m));
        l = m + __logf(sum) - xp;
      }
      if (isfinite(l)) lsum += l;
    }
    out[R + row] = lsum * (1.0f / TPOS);
  }
}

extern "C" void kernel_launch(void* const* d_in, const int* in_sizes, int n_in,
                              void* d_out, int out_size, void* d_ws, size_t ws_size,
                              hipStream_t stream) {
  (void)in_sizes; (void)n_in; (void)out_size; (void)ws_size;
  const float* old_feat = (const float*)d_in[0]; (void)old_feat; // unused ('nn' pair)
  const float* new_feat = (const float*)d_in[1];
  const int*   target   = (const int*)d_in[2];
  float*       out      = (float*)d_out;
  _Float16*    nf       = (_Float16*)d_ws;  // 8 MB f16 normalized feats

  cl_normalize_kernel<<<BSZ / WAVES, 256, 0, stream>>>(new_feat, nf);
  cl_mine_loss_kernel<<<BSZ / (WAVES * 16), 256, 0, stream>>>(nf, target, out);
}
